// BERT_41291815584151
// MI455X (gfx1250) — compile-verified, matrix operand order checked by reading, not run on hardware
//
#include <hip/hip_runtime.h>

typedef __bf16 bf16;
typedef __attribute__((ext_vector_type(8)))  float v8f;
typedef __attribute__((ext_vector_type(8)))  bf16  v8bf;
typedef __attribute__((ext_vector_type(16))) bf16  v16bf;

// ---------------------------------------------------------------------------
// WMMA helpers (layouts per CDNA5 ISA, bf16 A 16x32 / B 32x16)
// ---------------------------------------------------------------------------
__device__ __forceinline__ v16bf cat16(v8bf lo, v8bf hi) {
  return __builtin_shufflevector(lo, hi, 0,1,2,3,4,5,6,7,8,9,10,11,12,13,14,15);
}
// p points at &row[kh*8]; elements e0..7 = K kh*8.., e8..15 = K 16+kh*8..
__device__ __forceinline__ v16bf frag_ld(const bf16* p) {
  v8bf lo = *reinterpret_cast<const v8bf*>(p);
  v8bf hi = *reinterpret_cast<const v8bf*>(p + 16);
  return cat16(lo, hi);
}
__device__ __forceinline__ v8f wmma_bf16(v16bf a, v16bf b, v8f c) {
  return __builtin_amdgcn_wmma_f32_16x16x32_bf16(false, a, false, b, (short)0, c,
                                                 false, false);
}

// ---------------------------------------------------------------------------
// Async global->LDS copy (CDNA5, ASYNCcnt tracked). 16 bytes per lane.
// LDS operand = wave-relative offset = low 32 bits of the generic address.
// ---------------------------------------------------------------------------
__device__ __forceinline__ void async_ld16(const bf16* g, bf16* l) {
  asm volatile("global_load_async_to_lds_b128 %0, %1, off"
               :
               : "v"((unsigned)(size_t)l), "v"((unsigned long long)(size_t)g)
               : "memory");
}
__device__ __forceinline__ void wait_async0() {
  asm volatile("s_wait_asynccnt 0x0" ::: "memory");
}

// ---------------------------------------------------------------------------
// f32 -> bf16 weight conversion (vectorized, n multiple of 8)
// ---------------------------------------------------------------------------
__global__ void cvt_kernel(const float* __restrict__ src, bf16* __restrict__ dst,
                           long long n) {
  long long i = ((long long)blockIdx.x * blockDim.x + threadIdx.x) * 8;
  if (i < n) {
    const float4* s4 = reinterpret_cast<const float4*>(src + i);
    float4 a = s4[0], b = s4[1];
    v8bf o;
    o[0] = (bf16)a.x; o[1] = (bf16)a.y; o[2] = (bf16)a.z; o[3] = (bf16)a.w;
    o[4] = (bf16)b.x; o[5] = (bf16)b.y; o[6] = (bf16)b.z; o[7] = (bf16)b.w;
    *reinterpret_cast<v8bf*>(dst + i) = o;
  }
}

// ---------------------------------------------------------------------------
// embedding gather: out[row, 0:256] = bf16(token_emb[tokens[row], 0:256])
// ---------------------------------------------------------------------------
__global__ void gather_kernel(const int* __restrict__ tokens,
                              const float* __restrict__ emb,
                              bf16* __restrict__ out) {
  int row = blockIdx.x;
  int t = tokens[row];
  const float* s = emb + (size_t)t * 256;
  bf16* d = out + (size_t)row * 256;
  for (int c = threadIdx.x; c < 256; c += blockDim.x) d[c] = (bf16)s[c];
}

// ---------------------------------------------------------------------------
// Generic GEMM:  C[M,N] = act(A[M,K] @ W[K,N] + bias)
// bf16 in, f32 accum. 128x128 block tile, 8 waves (4x2), 32x64 wave tile.
// Double-buffered LDS; A staged via async-to-LDS, B transposed via VGPRs.
// ACT: 0=none, 1=leaky_relu, 2=gelu_tanh (branch-free sigmoid form).
// ---------------------------------------------------------------------------
template <int ACT, bool OF, bool OB>
__global__ __launch_bounds__(256) void gemm_kernel(
    const bf16* __restrict__ A, const bf16* __restrict__ Bw,
    const float* __restrict__ bias, float* __restrict__ outF,
    bf16* __restrict__ outB, int N, int K) {
  __shared__ __align__(16) bf16 As[2][128 * 40];  // [m][k], stride 40
  __shared__ __align__(16) bf16 Bs[2][128 * 40];  // [n][k] (transposed)
  int tid  = threadIdx.x;
  int wave = tid >> 5, lane = tid & 31;
  int lq = lane & 15, kh = lane >> 4;
  int wm = wave & 3, wn = wave >> 2;  // 4x2 wave grid
  int m0 = blockIdx.y * 128, n0 = blockIdx.x * 128;

  // per-thread staging coordinates
  const int arow = tid >> 1, ac0 = (tid & 1) * 16;      // A tile 128x32
  const int nb0 = (tid & 15) * 8, kr = tid >> 4;        // B tile 32x128

  v8f acc[2][4];
#pragma unroll
  for (int mt = 0; mt < 2; ++mt)
#pragma unroll
    for (int nt = 0; nt < 4; ++nt)
#pragma unroll
      for (int e = 0; e < 8; ++e) acc[mt][nt][e] = 0.0f;

  const bf16* aRow = A + (size_t)(m0 + arow) * K;
  uint4 bst0, bst1;

  // prologue: stage tile 0
  async_ld16(aRow + ac0, &As[0][arow * 40 + ac0]);
  async_ld16(aRow + ac0 + 8, &As[0][arow * 40 + ac0 + 8]);
  bst0 = *reinterpret_cast<const uint4*>(Bw + (size_t)kr * N + n0 + nb0);
  bst1 = *reinterpret_cast<const uint4*>(Bw + (size_t)(kr + 16) * N + n0 + nb0);
  {
    const bf16* p0 = reinterpret_cast<const bf16*>(&bst0);
    const bf16* p1 = reinterpret_cast<const bf16*>(&bst1);
#pragma unroll
    for (int i = 0; i < 8; ++i) {
      Bs[0][(nb0 + i) * 40 + kr] = p0[i];
      Bs[0][(nb0 + i) * 40 + kr + 16] = p1[i];
    }
  }

  const int nIter = K >> 5;
  for (int it = 0; it < nIter; ++it) {
    const int cur = it & 1;
    wait_async0();
    __syncthreads();  // tile `cur` (async A + ds-stored B) visible to all

    const bool nxt = (it + 1) < nIter;
    if (nxt) {
      const int k0 = (it + 1) * 32;
      async_ld16(aRow + k0 + ac0, &As[cur ^ 1][arow * 40 + ac0]);
      async_ld16(aRow + k0 + ac0 + 8, &As[cur ^ 1][arow * 40 + ac0 + 8]);
      bst0 = *reinterpret_cast<const uint4*>(Bw + (size_t)(k0 + kr) * N + n0 + nb0);
      bst1 = *reinterpret_cast<const uint4*>(Bw + (size_t)(k0 + kr + 16) * N + n0 + nb0);
    }

    v16bf af[2], bfr[4];
#pragma unroll
    for (int mt = 0; mt < 2; ++mt)
      af[mt] = frag_ld(&As[cur][(wm * 32 + mt * 16 + lq) * 40 + kh * 8]);
#pragma unroll
    for (int nt = 0; nt < 4; ++nt)
      bfr[nt] = frag_ld(&Bs[cur][(wn * 64 + nt * 16 + lq) * 40 + kh * 8]);
#pragma unroll
    for (int mt = 0; mt < 2; ++mt)
#pragma unroll
      for (int nt = 0; nt < 4; ++nt)
        acc[mt][nt] = wmma_bf16(af[mt], bfr[nt], acc[mt][nt]);

    if (nxt) {  // scatter B into the other buffer while WMMAs drain
      const bf16* p0 = reinterpret_cast<const bf16*>(&bst0);
      const bf16* p1 = reinterpret_cast<const bf16*>(&bst1);
#pragma unroll
      for (int i = 0; i < 8; ++i) {
        Bs[cur ^ 1][(nb0 + i) * 40 + kr] = p0[i];
        Bs[cur ^ 1][(nb0 + i) * 40 + kr + 16] = p1[i];
      }
    }
  }

  // epilogue: C layout => row = r + 8*kh, col = lq (within each 16x16 tile)
#pragma unroll
  for (int mt = 0; mt < 2; ++mt) {
#pragma unroll
    for (int nt = 0; nt < 4; ++nt) {
      int col = n0 + wn * 64 + nt * 16 + lq;
      float bv = bias[col];
      size_t idx = (size_t)(m0 + wm * 32 + mt * 16 + 8 * kh) * N + col;
#pragma unroll
      for (int r = 0; r < 8; ++r) {
        float v = acc[mt][nt][r] + bv;
        if (ACT == 1) {
          v = (v >= 0.0f) ? v : 0.01f * v;
        } else if (ACT == 2) {
          // 0.5*x*(1+tanh(y)) == x * sigmoid(2y), branch-free
          float y = 0.7978845608028654f * (v + 0.044715f * v * v * v);
          v = v / (1.0f + __expf(-2.0f * y));
        }
        if (OF) outF[idx] = v;
        if (OB) outB[idx] = (bf16)v;
        idx += (size_t)N;
      }
    }
  }
}

// ---------------------------------------------------------------------------
// LayerNorm with fused residual. mode 0: torch (biased var, eps in sqrt, 1e-5)
//                               mode 1: custom (ddof=1 std, eps on std, 1e-6)
// res_bcast: 1 -> res indexed by (row % 512) (positional encoding)
// ---------------------------------------------------------------------------
__device__ __forceinline__ float block_sum(float v, float* sbuf) {
#pragma unroll
  for (int o = 16; o > 0; o >>= 1) v += __shfl_xor(v, o, 32);
  if ((threadIdx.x & 31) == 0) sbuf[threadIdx.x >> 5] = v;
  __syncthreads();
  float t = (threadIdx.x < 8) ? sbuf[threadIdx.x] : 0.0f;
#pragma unroll
  for (int o = 4; o > 0; o >>= 1) t += __shfl_xor(t, o, 32);
  if (threadIdx.x == 0) sbuf[0] = t;
  __syncthreads();
  float r = sbuf[0];
  __syncthreads();
  return r;
}

__global__ __launch_bounds__(256) void ln_kernel(
    const float* __restrict__ src, const float* __restrict__ res, int res_bcast,
    const float* __restrict__ g, const float* __restrict__ bta,
    float* __restrict__ outF, bf16* __restrict__ outB, int mode) {
  __shared__ float sbuf[8];
  int row = blockIdx.x;
  const float* s = src + (size_t)row * 768;
  const float* r = res + (size_t)(res_bcast ? (row & 511) : row) * 768;
  float v[3];
  float sum = 0.0f;
#pragma unroll
  for (int j = 0; j < 3; ++j) {
    int c = threadIdx.x + j * 256;
    v[j] = s[c] + r[c];
    sum += v[j];
  }
  float mean = block_sum(sum, sbuf) * (1.0f / 768.0f);
  float sq = 0.0f;
#pragma unroll
  for (int j = 0; j < 3; ++j) {
    float d = v[j] - mean;
    sq += d * d;
  }
  float ssq = block_sum(sq, sbuf);
  float inv;
  if (mode == 0) inv = rsqrtf(ssq * (1.0f / 768.0f) + 1e-5f);
  else           inv = 1.0f / (sqrtf(ssq * (1.0f / 767.0f)) + 1e-6f);
#pragma unroll
  for (int j = 0; j < 3; ++j) {
    int c = threadIdx.x + j * 256;
    float y = (v[j] - mean) * inv * g[c] + bta[c];
    size_t idx = (size_t)row * 768 + c;
    outF[idx] = y;
    outB[idx] = (bf16)y;
  }
}

// ---------------------------------------------------------------------------
// Flash attention. grid = (S/128, B*H), block 256 (8 waves).
// Wave w owns q rows [q0+16w, q0+16w+16): online softmax is wave-private.
// Q/K staged with async-to-LDS; V transposed through VGPRs.
// ---------------------------------------------------------------------------
__global__ __launch_bounds__(256) void attn_kernel(
    const bf16* __restrict__ Q, const bf16* __restrict__ K,
    const bf16* __restrict__ V, const int* __restrict__ tokens,
    bf16* __restrict__ O) {
  constexpr int S = 512, D = 768, H = 12;
  __shared__ __align__(16) bf16 Qs[128 * 72];     // [q][dk]
  __shared__ __align__(16) bf16 Ks[64 * 72];      // [key][dk]
  __shared__ __align__(16) bf16 Vts[64 * 72];     // [dk][key] (transposed)
  __shared__ __align__(16) bf16 Ps[8 * 16 * 72];  // per-wave P tile [q][key]
  __shared__ float maskv[64];

  int bh = blockIdx.y;
  int b = bh / H, h = bh % H;
  int q0 = blockIdx.x * 128;
  int tid = threadIdx.x;
  int wid = tid >> 5, lane = tid & 31;
  int lq = lane & 15, kh = lane >> 4;

  const bf16* qbase = Q + ((size_t)b * S) * D + h * 64;
  const bf16* kbase = K + ((size_t)b * S) * D + h * 64;
  const bf16* vbase = V + ((size_t)b * S) * D + h * 64;

  {  // Q tile 128x64 via async copy
    int row = tid >> 1, c0 = (tid & 1) * 32;
    const bf16* g = qbase + (size_t)(q0 + row) * D + c0;
    bf16* l = &Qs[row * 72 + c0];
    async_ld16(g, l);
    async_ld16(g + 8, l + 8);
    async_ld16(g + 16, l + 16);
    async_ld16(g + 24, l + 24);
  }

  float m_run[8], l_run[8];
  v8f o_acc[4];
#pragma unroll
  for (int r = 0; r < 8; ++r) { m_run[r] = -3.0e38f; l_run[r] = 0.0f; }
#pragma unroll
  for (int j = 0; j < 4; ++j)
#pragma unroll
    for (int e = 0; e < 8; ++e) o_acc[j][e] = 0.0f;

  const float scale = 0.125f;  // 1/sqrt(64)

  for (int kb = 0; kb < S / 64; ++kb) {
    __syncthreads();  // everyone done reading previous K/V
    {
      int row = tid >> 2, c0 = (tid & 3) * 16;
      const bf16* kg = kbase + (size_t)(kb * 64 + row) * D + c0;
      async_ld16(kg, &Ks[row * 72 + c0]);
      async_ld16(kg + 8, &Ks[row * 72 + c0 + 8]);
      // V block transposed through VGPRs
      const uint4* vg =
          reinterpret_cast<const uint4*>(vbase + (size_t)(kb * 64 + row) * D + c0);
      uint4 v0 = vg[0], v1 = vg[1];
      const bf16* vp0 = reinterpret_cast<const bf16*>(&v0);
      const bf16* vp1 = reinterpret_cast<const bf16*>(&v1);
#pragma unroll
      for (int i = 0; i < 8; ++i) {
        Vts[(c0 + i) * 72 + row] = vp0[i];
        Vts[(c0 + 8 + i) * 72 + row] = vp1[i];
      }
    }
    if (tid < 64) {
      int t = tokens[b * S + kb * 64 + tid];
      maskv[tid] = (t > 0) ? 0.0f : -1e9f;
    }
    wait_async0();
    __syncthreads();

    // scores: Q(16x64) @ K^T(64x64) for this wave's rows
    float sc[4][8];
#pragma unroll
    for (int nt = 0; nt < 4; ++nt) {
      v8f acc;
#pragma unroll
      for (int e = 0; e < 8; ++e) acc[e] = 0.0f;
#pragma unroll
      for (int ks = 0; ks < 64; ks += 32) {
        v16bf aq = frag_ld(&Qs[(wid * 16 + lq) * 72 + ks + kh * 8]);
        v16bf bk = frag_ld(&Ks[(nt * 16 + lq) * 72 + ks + kh * 8]);
        acc = wmma_bf16(aq, bk, acc);
      }
      float sm = maskv[nt * 16 + lq];
#pragma unroll
      for (int r = 0; r < 8; ++r) sc[nt][r] = acc[r] * scale + sm;
    }

    // online softmax per row (rows r + 8*kh; reduce over 16 lanes of a half)
    float corr[8];
#pragma unroll
    for (int r = 0; r < 8; ++r) {
      float mx = sc[0][r];
#pragma unroll
      for (int nt = 1; nt < 4; ++nt) mx = fmaxf(mx, sc[nt][r]);
#pragma unroll
      for (int o = 8; o > 0; o >>= 1) mx = fmaxf(mx, __shfl_xor(mx, o, 32));
      float nm = fmaxf(m_run[r], mx);
      corr[r] = __expf(m_run[r] - nm);
      m_run[r] = nm;
      float rs = 0.0f;
#pragma unroll
      for (int nt = 0; nt < 4; ++nt) {
        float p = __expf(sc[nt][r] - nm);
        sc[nt][r] = p;
        rs += p;
      }
#pragma unroll
      for (int o = 8; o > 0; o >>= 1) rs += __shfl_xor(rs, o, 32);
      l_run[r] = l_run[r] * corr[r] + rs;
    }

#pragma unroll
    for (int j = 0; j < 4; ++j)
#pragma unroll
      for (int r = 0; r < 8; ++r) o_acc[j][r] *= corr[r];

    // stage P (wave-private region, bf16) then O += P @ V
#pragma unroll
    for (int nt = 0; nt < 4; ++nt)
#pragma unroll
      for (int r = 0; r < 8; ++r)
        Ps[(wid * 16 + r + 8 * kh) * 72 + nt * 16 + lq] = (bf16)sc[nt][r];

#pragma unroll
    for (int j = 0; j < 4; ++j) {
#pragma unroll
      for (int kc = 0; kc < 64; kc += 32) {
        v16bf ap = frag_ld(&Ps[(wid * 16 + lq) * 72 + kc + kh * 8]);
        v16bf bv = frag_ld(&Vts[(j * 16 + lq) * 72 + kc + kh * 8]);
        o_acc[j] = wmma_bf16(ap, bv, o_acc[j]);
      }
    }
  }

#pragma unroll
  for (int r = 0; r < 8; ++r) {
    float inv = 1.0f / l_run[r];
    int qrow = q0 + wid * 16 + r + 8 * kh;
    size_t base = ((size_t)b * S + qrow) * D + h * 64;
#pragma unroll
    for (int j = 0; j < 4; ++j)
      O[base + j * 16 + lq] = (bf16)(o_acc[j][r] * inv);
  }
}

// ---------------------------------------------------------------------------
// Host orchestration
// ---------------------------------------------------------------------------
static void gemm_call(int act, const bf16* A, const bf16* W, const float* bias,
                      float* oF, bf16* oB, int Mm, int N, int K,
                      hipStream_t s) {
  dim3 g(N / 128, Mm / 128);
  if (oF) {
    if (act == 0)      gemm_kernel<0, true, false><<<g, 256, 0, s>>>(A, W, bias, oF, oB, N, K);
    else if (act == 1) gemm_kernel<1, true, false><<<g, 256, 0, s>>>(A, W, bias, oF, oB, N, K);
    else               gemm_kernel<2, true, false><<<g, 256, 0, s>>>(A, W, bias, oF, oB, N, K);
  } else {
    if (act == 0)      gemm_kernel<0, false, true><<<g, 256, 0, s>>>(A, W, bias, oF, oB, N, K);
    else if (act == 1) gemm_kernel<1, false, true><<<g, 256, 0, s>>>(A, W, bias, oF, oB, N, K);
    else               gemm_kernel<2, false, true><<<g, 256, 0, s>>>(A, W, bias, oF, oB, N, K);
  }
}

extern "C" void kernel_launch(void* const* d_in, const int* in_sizes, int n_in,
                              void* d_out, int out_size, void* d_ws,
                              size_t ws_size, hipStream_t stream) {
  (void)in_sizes; (void)n_in; (void)out_size; (void)ws_size;
  const int*   tokens    = (const int*)d_in[0];
  const float* token_emb = (const float*)d_in[1];
  const float* emb_w1    = (const float*)d_in[2];
  const float* emb_b1    = (const float*)d_in[3];
  const float* emb_w2    = (const float*)d_in[4];
  const float* emb_b2    = (const float*)d_in[5];
  const float* pe        = (const float*)d_in[6];
  const float* ln0_g     = (const float*)d_in[7];
  const float* ln0_b     = (const float*)d_in[8];
  const float* wq        = (const float*)d_in[9];
  const float* bq        = (const float*)d_in[10];
  const float* wk        = (const float*)d_in[11];
  const float* bk        = (const float*)d_in[12];
  const float* wvv       = (const float*)d_in[13];
  const float* bv        = (const float*)d_in[14];
  const float* wo        = (const float*)d_in[15];
  const float* bo        = (const float*)d_in[16];
  const float* ff_w1     = (const float*)d_in[17];
  const float* ff_b1     = (const float*)d_in[18];
  const float* ff_w2     = (const float*)d_in[19];
  const float* ff_b2     = (const float*)d_in[20];
  const float* ln1_g     = (const float*)d_in[21];
  const float* ln1_b     = (const float*)d_in[22];
  const float* ln2_g     = (const float*)d_in[23];
  const float* ln2_b     = (const float*)d_in[24];

  char* ws = (char*)d_ws;
  size_t off = 0;
  auto alloc = [&](size_t bytes) -> char* {
    char* p = ws + off;
    off = (off + bytes + 255) & ~(size_t)255;
    return p;
  };

  const size_t M = 16384;  // B*S
  bf16* w1b  = (bf16*)alloc(262144ull * 2);
  bf16* w2b  = (bf16*)alloc(786432ull * 2);
  bf16* wqb  = (bf16*)alloc(4718592ull * 2);
  bf16* wkb  = (bf16*)alloc(4718592ull * 2);
  bf16* wvb  = (bf16*)alloc(4718592ull * 2);
  bf16* wob  = (bf16*)alloc(4718592ull * 2);
  bf16* fw1b = (bf16*)alloc(18874368ull * 2);
  bf16* fw2b = (bf16*)alloc(18874368ull * 2);
  bf16* e0B  = (bf16*)alloc(M * 256 * 2);
  bf16* h1B  = (bf16*)alloc(M * 1024 * 2);
  bf16* ffB  = (bf16*)alloc(M * 3072 * 2);
  bf16* xB   = (bf16*)alloc(M * 768 * 2);
  bf16* qB   = (bf16*)alloc(M * 768 * 2);
  bf16* kB   = (bf16*)alloc(M * 768 * 2);
  bf16* vB   = (bf16*)alloc(M * 768 * 2);
  bf16* aoB  = (bf16*)alloc(M * 768 * 2);
  float* xF  = (float*)alloc(M * 768 * 4);
  float* tF  = (float*)alloc(M * 768 * 4);

  auto cvt = [&](const float* s, bf16* d, long long n) {
    int grid = (int)((n + 2047) / 2048);
    cvt_kernel<<<grid, 256, 0, stream>>>(s, d, n);
  };
  cvt(emb_w1, w1b, 262144);
  cvt(emb_w2, w2b, 786432);
  cvt(wq, wqb, 4718592);
  cvt(wk, wkb, 4718592);
  cvt(wvv, wvb, 4718592);
  cvt(wo, wob, 4718592);
  cvt(ff_w1, fw1b, 18874368);
  cvt(ff_w2, fw2b, 18874368);

  // embedding MLP + positional + torch LayerNorm
  gather_kernel<<<16384, 64, 0, stream>>>(tokens, token_emb, e0B);
  gemm_call(1, e0B, w1b, emb_b1, nullptr, h1B, 16384, 1024, 256, stream);
  gemm_call(1, h1B, w2b, emb_b2, tF, nullptr, 16384, 768, 1024, stream);
  ln_kernel<<<16384, 256, 0, stream>>>(tF, pe, 1, ln0_g, ln0_b, xF, xB, 0);

  for (int i = 0; i < 8; ++i) {
    const bf16* wq_i = wqb + (size_t)i * 589824;
    const bf16* wk_i = wkb + (size_t)i * 589824;
    const bf16* wv_i = wvb + (size_t)i * 589824;
    const bf16* wo_i = wob + (size_t)i * 589824;
    gemm_call(0, xB, wq_i, bq + i * 768, nullptr, qB, 16384, 768, 768, stream);
    gemm_call(0, xB, wk_i, bk + i * 768, nullptr, kB, 16384, 768, 768, stream);
    gemm_call(0, xB, wv_i, bv + i * 768, nullptr, vB, 16384, 768, 768, stream);
    attn_kernel<<<dim3(4, 384), 256, 0, stream>>>(qB, kB, vB, tokens, aoB);
    gemm_call(0, aoB, wo_i, bo + i * 768, tF, nullptr, 16384, 768, 768, stream);
    ln_kernel<<<16384, 256, 0, stream>>>(tF, xF, 0, ln1_g + i * 768,
                                         ln1_b + i * 768, xF, xB, 1);
    gemm_call(2, xB, fw1b + (size_t)i * 2359296, ff_b1 + i * 3072, nullptr, ffB,
              16384, 3072, 768, stream);
    gemm_call(0, ffB, fw2b + (size_t)i * 2359296, ff_b2 + i * 768, tF, nullptr,
              16384, 768, 3072, stream);
    float* xo = (i == 7) ? (float*)d_out : xF;
    ln_kernel<<<16384, 256, 0, stream>>>(tF, xF, 0, ln2_g + i * 768,
                                         ln2_b + i * 768, xo, xB, 1);
  }
}
